// EncoderLayer_61237643706837
// MI455X (gfx1250) — compile-verified, matrix operand order checked by reading, not run on hardware
//
#include <hip/hip_runtime.h>
#include <hip/hip_bf16.h>

// ---------------------------------------------------------------------------
// ViT encoder block for MI455X (gfx1250), bf16 WMMA pipeline.
// B=32, N=577 (padded to 608 for attention), E=768, H=12, Dh=64, MLP=3072.
// GEMMs: triple-buffered LDS, GLOBAL_LOAD_ASYNC_TO_LDS_B128 (ASYNCcnt),
// 2 tiles of prefetch in flight.
// ---------------------------------------------------------------------------

typedef __attribute__((ext_vector_type(16))) __bf16       v16bf;
typedef __attribute__((ext_vector_type(8)))  float        v8f;
typedef __attribute__((ext_vector_type(4)))  unsigned int u32x4;

static constexpr int Bn    = 32;
static constexpr int Ntok  = 577;
static constexpr int E     = 768;
static constexpr int Hh    = 12;
static constexpr int Dh    = 64;
static constexpr int MLPd  = 3072;
static constexpr int Mrows = Bn * Ntok;   // 18464
static constexpr int NPAD  = 608;         // 38*16, >= 577
static constexpr int BHn   = Bn * Hh;     // 384

struct Frag32 { u32x4 lo, hi; };

// Load a 16x32 bf16 A/B-style fragment half (caller folds in the per-lane
// (lane>=16 ? 8 : 0) K offset). Reads p[0..7] and p[16..23].
static __device__ inline v16bf ld_frag(const __bf16* p) {
    Frag32 f;
    f.lo = *(const u32x4*)(p);
    f.hi = *(const u32x4*)(p + 16);
    return __builtin_bit_cast(v16bf, f);
}

static __device__ inline v8f vzero8() {
    v8f z;
#pragma unroll
    for (int i = 0; i < 8; ++i) z[i] = 0.0f;
    return z;
}

static __device__ inline v8f wmma_bf16(v16bf a, v16bf b, v8f c) {
    return __builtin_amdgcn_wmma_f32_16x16x32_bf16(
        false, a, false, b, (short)0, c, false, false);
}

// ---- CDNA5 async global->LDS copy (16B per lane), tracked by ASYNCcnt -----
static __device__ inline void async_b128(void* lds, const void* gsrc) {
    unsigned           off = (unsigned)(uintptr_t)lds;   // low 32b = LDS offset
    unsigned long long ga  = (unsigned long long)(uintptr_t)gsrc;
    asm volatile("global_load_async_to_lds_b128 %0, %1, off"
                 :: "v"(off), "v"(ga) : "memory");
}
static __device__ inline void wait_async0() {
    asm volatile("s_wait_asynccnt 0x0" ::: "memory");
}
static __device__ inline void wait_async4() {
    asm volatile("s_wait_asynccnt 0x4" ::: "memory");
}
static __device__ inline void wait_async8() {
    asm volatile("s_wait_asynccnt 0x8" ::: "memory");
}

// ---------------------------------------------------------------------------
// Weight convert + transpose: dst[c*R + r] = bf16(src[r*C + c])
// ---------------------------------------------------------------------------
__global__ void transpose_bf16_kernel(const float* __restrict__ src,
                                      __bf16* __restrict__ dst, int R, int C) {
    int idx = blockIdx.x * 256 + threadIdx.x;
    if (idx >= R * C) return;
    int r = idx / C, c = idx - r * C;
    dst[(size_t)c * R + r] = (__bf16)src[(size_t)r * C + c];
}

// ---------------------------------------------------------------------------
// LayerNorm over 768, one wave32 per row, float4 loads, bf16x4 stores.
// ---------------------------------------------------------------------------
struct __align__(8) bf16x4 { __bf16 a, b, c, d; };

__global__ __launch_bounds__(256) void layernorm_bf16_kernel(
    const float* __restrict__ x, const float* __restrict__ g,
    const float* __restrict__ bb, __bf16* __restrict__ out, int rows) {
    int wid = threadIdx.x >> 5, lane = threadIdx.x & 31;
    int row = blockIdx.x * 8 + wid;
    if (row >= rows) return;
    const float4* xr4 = (const float4*)(x + (size_t)row * E);
    float4 v[6];
    float s = 0.f, ss = 0.f;
#pragma unroll
    for (int j = 0; j < 6; ++j) {
        v[j] = xr4[lane + j * 32];
        s  += v[j].x + v[j].y + v[j].z + v[j].w;
        ss += v[j].x * v[j].x + v[j].y * v[j].y + v[j].z * v[j].z + v[j].w * v[j].w;
    }
#pragma unroll
    for (int d = 16; d >= 1; d >>= 1) {
        s  += __shfl_xor(s,  d, 32);
        ss += __shfl_xor(ss, d, 32);
    }
    float mean = s * (1.0f / E);
    float var  = ss * (1.0f / E) - mean * mean;
    float rstd = rsqrtf(var + 1e-5f);
    const float4* g4 = (const float4*)g;
    const float4* b4 = (const float4*)bb;
    __bf16* orow = out + (size_t)row * E;
#pragma unroll
    for (int j = 0; j < 6; ++j) {
        int i4 = lane + j * 32;
        float4 gv = g4[i4], bv = b4[i4];
        bf16x4 o;
        o.a = (__bf16)((v[j].x - mean) * rstd * gv.x + bv.x);
        o.b = (__bf16)((v[j].y - mean) * rstd * gv.y + bv.y);
        o.c = (__bf16)((v[j].z - mean) * rstd * gv.z + bv.z);
        o.d = (__bf16)((v[j].w - mean) * rstd * gv.w + bv.w);
        *(bf16x4*)(orow + i4 * 4) = o;
    }
}

// ---------------------------------------------------------------------------
// bf16 GEMM: C[M,N] = A[M,K] @ Wt[N,K]^T, fused epilogues.
// Block tile 128x128, 8 waves (2 along M x 4 along N), wave tile 64x32
// (4x2 WMMA accumulators -> 8 wmma per K-step vs 6 fragment loads).
// Triple-buffered LDS, async global->LDS copies, 2 tiles in flight.
// EPI: 0=QKV split, 1=proj+gamma*resid, 2=FC1+GELU, 3=FC2+1e-5*resid
// ---------------------------------------------------------------------------
static constexpr int BM = 128, BNt = 128, BK = 32, LDT = 40, NBUF = 3;

template <int EPI>
__global__ __launch_bounds__(256) void gemm_bf16_kernel(
    const __bf16* __restrict__ A, const __bf16* __restrict__ Wt,
    int M, int K,
    const float* __restrict__ bias, const float* __restrict__ gamma,
    const float* __restrict__ resid, float* __restrict__ outF,
    __bf16* __restrict__ outB,
    __bf16* __restrict__ qb, __bf16* __restrict__ kb, __bf16* __restrict__ vtb) {
    __shared__ __bf16 At[NBUF][BM * LDT];   // 3 x 10,240 B
    __shared__ __bf16 Bt[NBUF][BNt * LDT];  // 3 x 10,240 B (total 61,440 B)

    const int tid  = threadIdx.x;
    const int lane = tid & 31;
    const int wid  = tid >> 5;
    const int wm   = (wid & 1) * 64;    // wave M offset in tile
    const int wn   = (wid >> 1) * 32;   // wave N offset in tile
    const int m0   = blockIdx.x * BM;
    const int n0   = blockIdx.y * BNt;

    v8f acc[4][2];
#pragma unroll
    for (int i = 0; i < 4; ++i)
#pragma unroll
        for (int j = 0; j < 2; ++j) acc[i][j] = vzero8();

    const int srow  = tid >> 1;            // 0..127 (rows of both tiles)
    const int shalf = (tid & 1) * 16;      // 0 or 16 elements
    const int off   = ((lane >> 4) & 1) * 8;
    const int r15   = lane & 15;
    const int kTiles = K / BK;

    // Stage tile kt into buffer buf: 4 async b128 per thread (A:2, B:2).
    auto stage = [&](int kt, int buf) {
        int gm  = m0 + srow;
        int gmc = gm < M ? gm : (M - 1);   // clamp: garbage rows never stored
        const __bf16* sa = A  + (size_t)gmc * K + kt * BK + shalf;
        const __bf16* sb = Wt + (size_t)(n0 + srow) * K + kt * BK + shalf;
        __bf16* da = &At[buf][srow * LDT + shalf];
        __bf16* db = &Bt[buf][srow * LDT + shalf];
        async_b128(da,     sa);
        async_b128(da + 8, sa + 8);
        async_b128(db,     sb);
        async_b128(db + 8, sb + 8);
    };

    stage(0, 0);
    stage(1, 1);
    for (int kt = 0; kt < kTiles; ++kt) {
        const int cur = kt % NBUF;
        if (kt + 2 < kTiles) {
            stage(kt + 2, (kt + 2) % NBUF);  // keep 2 tiles in flight
            wait_async8();   // oldest 4 copies (tile kt) done, in-order
        } else if (kt + 1 < kTiles) {
            wait_async4();
        } else {
            wait_async0();
        }
        __syncthreads();

        v16bf b0 = ld_frag(&Bt[cur][(wn +      r15) * LDT + off]);
        v16bf b1 = ld_frag(&Bt[cur][(wn + 16 + r15) * LDT + off]);
#pragma unroll
        for (int i = 0; i < 4; ++i) {
            v16bf a = ld_frag(&At[cur][(wm + i * 16 + r15) * LDT + off]);
            acc[i][0] = wmma_bf16(a, b0, acc[i][0]);
            acc[i][1] = wmma_bf16(a, b1, acc[i][1]);
        }
        __syncthreads();  // readers done before this buffer is refilled
    }

    // Epilogue: C layout: elem r -> M = r + (lane>=16 ? 8 : 0), N = lane&15
    const int mup = (lane >> 4) * 8;
#pragma unroll
    for (int ti = 0; ti < 4; ++ti) {
#pragma unroll
        for (int tj = 0; tj < 2; ++tj) {
            int mb = m0 + wm + ti * 16 + mup;
            int gn = n0 + wn + tj * 16 + r15;
#pragma unroll
            for (int r = 0; r < 8; ++r) {
                int gm = mb + r;
                if (gm >= M) continue;
                float val = acc[ti][tj][r];
                if (EPI == 0) {
                    // QKV split; Q pre-scaled by Dh^-0.5; V stored transposed.
                    int which = gn / E;
                    int rem   = gn - which * E;
                    int hh = rem >> 6, d = rem & 63;
                    int b_ = gm / Ntok, nn = gm - b_ * Ntok;
                    size_t bh = (size_t)(b_ * Hh + hh);
                    if (which == 0)
                        qb[(bh * NPAD + nn) * Dh + d] = (__bf16)(val * 0.125f);
                    else if (which == 1)
                        kb[(bh * NPAD + nn) * Dh + d] = (__bf16)val;
                    else
                        vtb[(bh * Dh + d) * NPAD + nn] = (__bf16)val;
                } else if (EPI == 1) {
                    float v = val + bias[gn];
                    size_t idx = (size_t)gm * E + gn;
                    outF[idx] = resid[idx] + v * gamma[gn];
                } else if (EPI == 2) {
                    float v = val + bias[gn];
                    float ge = 0.5f * v * (1.0f + erff(v * 0.70710678118654752f));
                    outB[(size_t)gm * MLPd + gn] = (__bf16)ge;
                } else {
                    float v = val + bias[gn];
                    size_t idx = (size_t)gm * E + gn;
                    outF[idx] = resid[idx] + v * 1e-5f;  // ls2 gamma = init value
                }
            }
        }
    }
}

// ---------------------------------------------------------------------------
// Attention: one wave per (bh, 16 Q rows). S tile in LDS, 2-lane-per-row
// branchless softmax with mask, bf16 P in separate LDS buffer, then P@V via
// WMMA with V^T fragments streamed from global.
// ---------------------------------------------------------------------------
__global__ __launch_bounds__(32) void attn_kernel(
    const __bf16* __restrict__ q, const __bf16* __restrict__ k,
    const __bf16* __restrict__ vT, __bf16* __restrict__ attn_out) {
    __shared__ float  S[16][NPAD];   // 38,912 B
    __shared__ __bf16 P[16][NPAD];   // 19,456 B
    __shared__ __bf16 Qs[16][Dh];    //  2,048 B  (total 60,416 B)

    const int lane = threadIdx.x;
    const int bh   = blockIdx.y;
    const int b_   = bh / Hh, h = bh - b_ * Hh;
    const int m0   = blockIdx.x * 16;
    const int off  = ((lane >> 4) & 1) * 8;
    const int r15  = lane & 15;
    const int mup  = (lane >> 4) * 8;

    {   // load Q tile (rows m0..m0+15, padded buffer -> no guard)
        int row = lane >> 1, col = (lane & 1) * 32;
        const __bf16* src = q + ((size_t)bh * NPAD + m0 + row) * Dh + col;
        __bf16* dst = &Qs[row][col];
#pragma unroll
        for (int c = 0; c < 4; ++c)
            ((u32x4*)dst)[c] = ((const u32x4*)src)[c];
    }

    // Pass 1: S = Q * K^T (Q already scaled)
    for (int jt = 0; jt < NPAD / 16; ++jt) {
        v8f acc = vzero8();
#pragma unroll
        for (int ks = 0; ks < Dh; ks += 32) {
            v16bf a  = ld_frag(&Qs[r15][ks + off]);
            v16bf bk = ld_frag(k + ((size_t)bh * NPAD + jt * 16 + r15) * Dh + ks + off);
            acc = wmma_bf16(a, bk, acc);
        }
#pragma unroll
        for (int r = 0; r < 8; ++r)
            S[mup + r][jt * 16 + r15] = acc[r];
    }

    // Softmax: 2 lanes per row; branchless ranges, mask cols >= 577.
    {
        float* Srow = &S[r15][0];
        const int half = lane >> 4;
        const int j0 = half * (NPAD / 2);        // 0 or 304
        const int jv = half ? Ntok : (NPAD / 2); // valid end: 304 or 577
        const int j1 = j0 + (NPAD / 2);
        float mx = -1e30f;
        for (int j = j0; j < jv; ++j) mx = fmaxf(mx, Srow[j]);
        mx = fmaxf(mx, __shfl_xor(mx, 16, 32));
        float s = 0.f;
        for (int j = j0; j < jv; ++j) {
            float e = __expf(Srow[j] - mx);
            Srow[j] = e; s += e;
        }
        s += __shfl_xor(s, 16, 32);
        float inv = 1.0f / s;
        for (int j = j0; j < jv; ++j) P[r15][j] = (__bf16)(Srow[j] * inv);
        for (int j = jv; j < j1; ++j) P[r15][j] = (__bf16)0.0f;  // half1 tail
    }
    // single wave: LDS ops are in-order, no barrier needed

    // Pass 2: out = P @ V  (V^T rows are contiguous)
#pragma unroll
    for (int nt = 0; nt < 4; ++nt) {
        v8f acc = vzero8();
        for (int kt = 0; kt < NPAD / 32; ++kt) {
            v16bf a  = ld_frag(&P[r15][kt * 32 + off]);
            v16bf bv = ld_frag(vT + ((size_t)bh * Dh + nt * 16 + r15) * NPAD + kt * 32 + off);
            acc = wmma_bf16(a, bv, acc);
        }
#pragma unroll
        for (int r = 0; r < 8; ++r) {
            int m = m0 + mup + r;
            if (m < Ntok)
                attn_out[((size_t)(b_ * Ntok + m)) * E + h * Dh + nt * 16 + r15] =
                    (__bf16)acc[r];
        }
    }
}

// ---------------------------------------------------------------------------
// Host driver
// ---------------------------------------------------------------------------
extern "C" void kernel_launch(void* const* d_in, const int* in_sizes, int n_in,
                              void* d_out, int out_size, void* d_ws, size_t ws_size,
                              hipStream_t stream) {
    (void)in_sizes; (void)n_in; (void)out_size; (void)ws_size;

    const float* x      = (const float*)d_in[0];
    const float* ln1_g  = (const float*)d_in[1];
    const float* ln1_b  = (const float*)d_in[2];
    const float* w_qkv  = (const float*)d_in[3];
    const float* w_proj = (const float*)d_in[4];
    const float* b_proj = (const float*)d_in[5];
    const float* ls1_g  = (const float*)d_in[6];
    const float* ln2_g  = (const float*)d_in[7];
    const float* ln2_b  = (const float*)d_in[8];
    const float* w_fc1  = (const float*)d_in[9];
    const float* b_fc1  = (const float*)d_in[10];
    const float* w_fc2  = (const float*)d_in[11];
    const float* b_fc2  = (const float*)d_in[12];
    float* out = (float*)d_out;

    char* ws = (char*)d_ws;
    size_t cur = 0;
    auto alloc = [&](size_t bytes) -> char* {
        char* p = ws + cur;
        cur = (cur + bytes + 255) & ~(size_t)255;
        return p;
    };
    const size_t SZ_QKV1 = (size_t)BHn * NPAD * Dh * 2;  // 29,884,416 B each

    __bf16* xln1   = (__bf16*)alloc((size_t)Mrows * E * 2);
    __bf16* wqkvT  = (__bf16*)alloc((size_t)(3 * E) * E * 2);
    __bf16* wprojT = (__bf16*)alloc((size_t)E * E * 2);
    __bf16* wfc1T  = (__bf16*)alloc((size_t)MLPd * E * 2);
    __bf16* wfc2T  = (__bf16*)alloc((size_t)E * MLPd * 2);
    __bf16* qb     = (__bf16*)alloc(SZ_QKV1);
    __bf16* kb     = (__bf16*)alloc(SZ_QKV1);
    __bf16* vtb    = (__bf16*)alloc(SZ_QKV1);
    __bf16* attn   = (__bf16*)alloc((size_t)Mrows * E * 2);
    float*  x1     = (float*) alloc((size_t)Mrows * E * 4);
    __bf16* xln2   = (__bf16*)alloc((size_t)Mrows * E * 2);
    __bf16* mact   = (__bf16*)alloc((size_t)Mrows * MLPd * 2);

    // Weight convert+transpose -> [N, K] bf16
    auto tgrid = [](int n) { return dim3((unsigned)((n + 255) / 256)); };
    transpose_bf16_kernel<<<tgrid(E * 3 * E), 256, 0, stream>>>(w_qkv,  wqkvT,  E, 3 * E);
    transpose_bf16_kernel<<<tgrid(E * E),     256, 0, stream>>>(w_proj, wprojT, E, E);
    transpose_bf16_kernel<<<tgrid(E * MLPd),  256, 0, stream>>>(w_fc1,  wfc1T,  E, MLPd);
    transpose_bf16_kernel<<<tgrid(MLPd * E),  256, 0, stream>>>(w_fc2,  wfc2T,  MLPd, E);

    // Zero-fill padded Q/K/V^T (rows 577..607 must be zero)
    hipMemsetAsync(qb,  0, SZ_QKV1, stream);
    hipMemsetAsync(kb,  0, SZ_QKV1, stream);
    hipMemsetAsync(vtb, 0, SZ_QKV1, stream);

    const int lnGrid = (Mrows + 7) / 8;
    const int tilesM = (Mrows + BM - 1) / BM;  // 145

    // LN1
    layernorm_bf16_kernel<<<lnGrid, 256, 0, stream>>>(x, ln1_g, ln1_b, xln1, Mrows);

    // QKV GEMM (no bias in reference): [18464,768] @ [768,2304]
    gemm_bf16_kernel<0><<<dim3(tilesM, (3 * E) / BNt), 256, 0, stream>>>(
        xln1, wqkvT, Mrows, E, nullptr, nullptr, nullptr, nullptr, nullptr,
        qb, kb, vtb);

    // Attention
    attn_kernel<<<dim3((Ntok + 15) / 16, BHn), 32, 0, stream>>>(qb, kb, vtb, attn);

    // Proj + bias, x1 = x + ls1_gamma * h
    gemm_bf16_kernel<1><<<dim3(tilesM, E / BNt), 256, 0, stream>>>(
        attn, wprojT, Mrows, E, b_proj, ls1_g, x, x1, nullptr,
        nullptr, nullptr, nullptr);

    // LN2
    layernorm_bf16_kernel<<<lnGrid, 256, 0, stream>>>(x1, ln2_g, ln2_b, xln2, Mrows);

    // FC1 + bias + exact GELU
    gemm_bf16_kernel<2><<<dim3(tilesM, MLPd / BNt), 256, 0, stream>>>(
        xln2, wfc1T, Mrows, E, b_fc1, nullptr, nullptr, nullptr, mact,
        nullptr, nullptr, nullptr);

    // FC2 + bias, out = x1 + 1e-5 * mlp
    gemm_bf16_kernel<3><<<dim3(tilesM, E / BNt), 256, 0, stream>>>(
        mact, wfc2T, Mrows, MLPd, b_fc2, nullptr, x1, out, nullptr,
        nullptr, nullptr, nullptr);
}
